// FeedForwardQuantum_65481071395349
// MI455X (gfx1250) — matrix operand order repaired and checked by reading, not compiled, not run on hardware
//
#include <hip/hip_runtime.h>
#include <hip/hip_bf16.h>
#include <math.h>

typedef __attribute__((ext_vector_type(16))) _Float16 v16h;
typedef __attribute__((ext_vector_type(2)))  _Float16 v2h;
typedef __attribute__((ext_vector_type(8)))  float    v8f;

#define EE 512      // embed
#define FF 2048     // ffn
#define QQ 8        // qubits
#define NTOK 32768  // B*S
#define MTILE 64    // tokens per workgroup
#define KC 32       // K chunk (WMMA K)
#define NKC (FF / KC)    // 64 chunks
#define NT_TOT (EE / 16) // 32 N-tiles
#define FRAG_H (32 * 16) // halfs per B fragment (32 lanes x 16)

// ---------------------------------------------------------------------------
// Prep: W2 [E=512, F=2048] f32 row-major  ->  f16 B-fragments.
// Fragment (nt, kc): lane l holds 16 contiguous halfs:
//   W2[nt*16 + l%16][kc*32 + 16*(l/16) + j], j=0..15
// stored at w2s + ((nt*NKC + kc)*32 + l)*16 halfs (32B per lane, coalesced).
// Contiguous k order (k = 16*(l/16) + j), also used for the A fragments in LDS.
// ---------------------------------------------------------------------------
__global__ __launch_bounds__(256) void ffq_prep_w2(const float* __restrict__ W2,
                                                   _Float16* __restrict__ w2s) {
  int idx  = blockIdx.x * blockDim.x + threadIdx.x;   // 0 .. 32*64*32-1
  int lane = idx & 31;
  int kc   = (idx >> 5) & (NKC - 1);
  int nt   = idx >> 11;                               // 0..31
  int n    = nt * 16 + (lane & 15);
  int kb   = kc * 32 + ((lane >> 4) << 4);
  const float* src = W2 + (size_t)n * FF + kb;        // 64B aligned
  float4 s0 = ((const float4*)src)[0];
  float4 s1 = ((const float4*)src)[1];
  float4 s2 = ((const float4*)src)[2];
  float4 s3 = ((const float4*)src)[3];
  float sv[16] = {s0.x,s0.y,s0.z,s0.w, s1.x,s1.y,s1.z,s1.w,
                  s2.x,s2.y,s2.z,s2.w, s3.x,s3.y,s3.z,s3.w};
  v16h v;
#pragma unroll
  for (int j = 0; j < 16; ++j) v[j] = (_Float16)sv[j];
  *(v16h*)(w2s + (size_t)idx * 16) = v;
}

// ---------------------------------------------------------------------------
// Fused main kernel: 512 threads (16 waves), 64 tokens per workgroup.
// Waves: wm = wid&3 (16-token M-tile), wn = wid>>2 (128-column N-group).
// Per K-chunk, in issue order:
//   [W1/b1 loads] -> [8 B-fragment loads] -> [prefetch next B] ->
//   [h FMAs + relu + cvt + LDS scatter] -> barrier -> [A ds_load] -> 8 WMMAs.
// B loads are issued ~barrier+40 VALU ops before their WMMA consumers.
// ---------------------------------------------------------------------------
__global__ __launch_bounds__(512) void ffq_main(const float* __restrict__ x,
                                                const float* __restrict__ theta,
                                                const float* __restrict__ W1,
                                                const float* __restrict__ b1,
                                                const float* __restrict__ b2,
                                                const _Float16* __restrict__ w2s,
                                                float* __restrict__ out) {
  __shared__ float qbuf[MTILE][QQ];                                    // 2 KB
  __shared__ _Float16 abuf[2][4][32][16] __attribute__((aligned(32))); // 8 KB

  const int t    = threadIdx.x;
  const int tok0 = blockIdx.x * MTILE;

  // ---- phase 0: quantum layer  q[m][i] = cos(x[tok,i]) * cos(theta[i]) ----
  {
    int tl = t >> 3;           // local token 0..63
    int qi = t & 7;
    float xv = x[(size_t)(tok0 + tl) * EE + qi];
    qbuf[tl][qi] = __cosf(xv) * __cosf(theta[qi]);
  }
  __syncthreads();

  // ---- producer assignment: thread -> (k pair, 2 token rows) ----
  const int kk2 = t & 15;        // k pair index: kk = 2*kk2 (0..30)
  const int m2  = t >> 4;        // 0..31 -> rows 2*m2, 2*m2+1
  const int kk  = kk2 * 2;
  const int ma  = 2 * m2, mb = 2 * m2 + 1;
  float qa[QQ], qb[QQ];
#pragma unroll
  for (int i = 0; i < QQ; ++i) { qa[i] = qbuf[ma][i]; qb[i] = qbuf[mb][i]; }
  // position of (kk, kk+1) inside the per-lane 16-half A fragment
  int hfk, pos;
  hfk = kk >> 4;
  pos = kk & 15;
  _Float16* dstA = &abuf[0][ma >> 4][(hfk << 4) | (ma & 15)][pos];
  _Float16* dstB = &abuf[0][mb >> 4][(hfk << 4) | (mb & 15)][pos];
  const int bufStrideH = 4 * 32 * 16;  // halfs per LDS A buffer

  // ---- consumer assignment ----
  const int lane = t & 31;
  const int wid  = t >> 5;      // 0..15
  const int wm   = wid & 3;     // M-tile within workgroup
  const int wn   = wid >> 2;    // 0..3 -> columns [wn*128, wn*128+128)

  v8f acc[8];
#pragma unroll
  for (int j = 0; j < 8; ++j) acc[j] = (v8f){};

  const _Float16* __restrict__ bbase =
      w2s + (((size_t)(wn * 8) * NKC) * 32 + lane) * 16;

#pragma unroll 1
  for (int it = 0; it < NKC; ++it) {
    const int k0  = it * KC;
    const int buf = it & 1;

    // ---- (1) producer operand loads FIRST (so waiting on them does not
    //      drain the B loads issued just after; LOADcnt retires in order) ----
    const float4* wp = (const float4*)(W1 + (size_t)(k0 + kk) * QQ); // 64B aligned
    float4 w0 = wp[0], w1 = wp[1], w2 = wp[2], w3 = wp[3];
    float2 bb = *(const float2*)(b1 + k0 + kk);

    // ---- (2) issue all 8 B-fragment loads for THIS iteration ----
    const _Float16* __restrict__ bp = bbase + (size_t)it * FRAG_H;
    v16h bf0 = *(const v16h*)(bp + (size_t)0 * NKC * FRAG_H);
    v16h bf1 = *(const v16h*)(bp + (size_t)1 * NKC * FRAG_H);
    v16h bf2 = *(const v16h*)(bp + (size_t)2 * NKC * FRAG_H);
    v16h bf3 = *(const v16h*)(bp + (size_t)3 * NKC * FRAG_H);
    v16h bf4 = *(const v16h*)(bp + (size_t)4 * NKC * FRAG_H);
    v16h bf5 = *(const v16h*)(bp + (size_t)5 * NKC * FRAG_H);
    v16h bf6 = *(const v16h*)(bp + (size_t)6 * NKC * FRAG_H);
    v16h bf7 = *(const v16h*)(bp + (size_t)7 * NKC * FRAG_H);

    // ---- (3) warm WGP$ for next iteration's fragments (untracked) ----
    if (it + 1 < NKC) {
      const _Float16* np = bp + FRAG_H;
#pragma unroll
      for (int j = 0; j < 8; ++j)
        __builtin_prefetch(np + (size_t)j * NKC * FRAG_H, 0, 3);
    }

    // ---- (4) produce h chunk: h[m][k0+kk .. +1] for m = ma, mb ----
    float wv[16] = {w0.x,w0.y,w0.z,w0.w, w1.x,w1.y,w1.z,w1.w,
                    w2.x,w2.y,w2.z,w2.w, w3.x,w3.y,w3.z,w3.w};
    float ha0 = bb.x, ha1 = bb.y, hb0 = bb.x, hb1 = bb.y;
#pragma unroll
    for (int j = 0; j < QQ; ++j) {
      ha0 = fmaf(qa[j], wv[j],     ha0);
      ha1 = fmaf(qa[j], wv[8 + j], ha1);
      hb0 = fmaf(qb[j], wv[j],     hb0);
      hb1 = fmaf(qb[j], wv[8 + j], hb1);
    }
    ha0 = fmaxf(ha0, 0.f); ha1 = fmaxf(ha1, 0.f);
    hb0 = fmaxf(hb0, 0.f); hb1 = fmaxf(hb1, 0.f);
    v2h pa; pa.x = (_Float16)ha0; pa.y = (_Float16)ha1;
    v2h pb; pb.x = (_Float16)hb0; pb.y = (_Float16)hb1;
    *(v2h*)(dstA + buf * bufStrideH) = pa;   // 4B-aligned LDS store
    *(v2h*)(dstB + buf * bufStrideH) = pb;

    __syncthreads();  // writes of buf visible; also fences reads from 2 iters ago

    // ---- (5) A from LDS, then the 8-WMMA chain (B loads drain here) ----
    v16h a = *(const v16h*)&abuf[buf][wm][lane][0];

#define WMMA_STEP(i, bfr) \
    acc[i] = __builtin_amdgcn_wmma_f32_16x16x32_f16( \
        false, a, false, (bfr), (short)0, acc[i], false, false)
    WMMA_STEP(0, bf0); WMMA_STEP(1, bf1); WMMA_STEP(2, bf2); WMMA_STEP(3, bf3);
    WMMA_STEP(4, bf4); WMMA_STEP(5, bf5); WMMA_STEP(6, bf6); WMMA_STEP(7, bf7);
#undef WMMA_STEP
  }

  // ---- epilogue: out[m][n] = acc + b2[n] ----
  const int hf = lane >> 4;     // C/D layout: row = 8*hf + r, col = lane%16
  const int lm = lane & 15;
#pragma unroll
  for (int j = 0; j < 8; ++j) {
    int col = wn * 128 + j * 16 + lm;
    float bias = b2[col];
    size_t rowbase = (size_t)(tok0 + wm * 16 + hf * 8) * EE + col;
#pragma unroll
    for (int r = 0; r < 8; ++r) {
      out[rowbase + (size_t)r * EE] = acc[j][r] + bias;
    }
  }
}

// ---------------------------------------------------------------------------
extern "C" void kernel_launch(void* const* d_in, const int* in_sizes, int n_in,
                              void* d_out, int out_size, void* d_ws, size_t ws_size,
                              hipStream_t stream) {
  const float* x     = (const float*)d_in[0];
  const float* theta = (const float*)d_in[1];
  const float* W1    = (const float*)d_in[2];
  const float* b1    = (const float*)d_in[3];
  const float* W2    = (const float*)d_in[4];
  const float* b2    = (const float*)d_in[5];
  float* out         = (float*)d_out;
  _Float16* w2s      = (_Float16*)d_ws;   // needs 2 MiB

  // 1) swizzle W2 into f16 WMMA B-fragments (L2-resident, read every iter)
  ffq_prep_w2<<<(NT_TOT * NKC * 32) / 256, 256, 0, stream>>>(W2, w2s);
  // 2) fused quantum-layer + GEMM1 + relu + GEMM2 + bias
  ffq_main<<<NTOK / MTILE, 512, 0, stream>>>(x, theta, W1, b1, b2, w2s, out);
}
